// MaskedAttention_77558519431616
// MI455X (gfx1250) — matrix operand order repaired and checked by reading, not compiled, not run on hardware
//
#include <hip/hip_runtime.h>

// ---------------- problem constants ----------------
constexpr int Bs = 4;
constexpr int Ts = 2048;
constexpr int Ds = 1024;

// ---------------- vector types (plain clang ext vectors; union-safe) ----------
typedef __attribute__((ext_vector_type(16))) __bf16        v16bf;
typedef __attribute__((ext_vector_type(8)))  float         v8f;
typedef __attribute__((ext_vector_type(4)))  unsigned int  u32x4;
typedef __attribute__((ext_vector_type(8)))  int           i32x8;
typedef __attribute__((ext_vector_type(4)))  int           i32x4;
typedef __attribute__((ext_vector_type(4)))  float         f32x4;

union FragCast { u32x4 q[2]; v16bf v; };

#if defined(__has_builtin)
#if __has_builtin(__builtin_amdgcn_tensor_load_to_lds)
#define HAVE_TDM 1
#endif
#endif
#ifndef HAVE_TDM
#define HAVE_TDM 0
#endif

// fp32 -> bf16 round-to-nearest-even
__device__ __forceinline__ unsigned short f2bf(float x) {
    unsigned int u = __builtin_bit_cast(unsigned int, x);
    unsigned int r = u + 0x7FFFu + ((u >> 16) & 1u);
    return (unsigned short)(r >> 16);
}

// Tiles in LDS are [rows][64] bf16 (row stride 64 elements = 128 B).
// A fragment (16x32, MxK): lane = 16*h + r holds row r,
// elements 0..7 -> K = kk+8h.. ; elements 8..15 -> K = kk+16+8h..
__device__ __forceinline__ v16bf load_fragA(const unsigned short* tile, int row, int kk, int h) {
    const unsigned short* p = tile + row * 64 + kk;
    FragCast f;
    f.q[0] = *(const u32x4*)(p + h * 8);
    f.q[1] = *(const u32x4*)(p + 16 + h * 8);
    return f.v;
}

// B fragment (32x16, KxN) staged as [N][K] tile: lane = 16*h + n holds
// column n, elements 0..7 -> K = kk+8h.. ; elements 8..15 -> K = kk+16+8h..
__device__ __forceinline__ v16bf load_fragB(const unsigned short* tile, int col, int kk, int h) {
    const unsigned short* p = tile + col * 64 + kk;
    FragCast f;
    f.q[0] = *(const u32x4*)(p + h * 8);
    f.q[1] = *(const u32x4*)(p + 16 + h * 8);
    return f.v;
}

__device__ __forceinline__ v8f wmma_bf16(v16bf a, v16bf b, v8f c) {
    return __builtin_amdgcn_wmma_f32_16x16x32_bf16(false, a, false, b, (short)0, c, false, false);
}

// ---------------------------------------------------------------------
// Tensor Data Mover: 2-D bf16 tile (tile_d0 contiguous elems x tile_d1
// rows, source row stride = stride_elems) -> dense LDS tile.
// D# packing per CDNA5 ISA (groups 2/3 zero).
// ---------------------------------------------------------------------
__device__ __forceinline__ void tdm_load_2d(unsigned int lds_off, const void* gptr,
                                            unsigned int tile_d0, unsigned int tile_d1,
                                            unsigned int stride_elems) {
#if HAVE_TDM
    unsigned long long ga = (unsigned long long)gptr;
    u32x4 g0;
    g0.x = 1u;                                       // count=1, user descriptor
    g0.y = lds_off;                                  // lds_addr [63:32]
    g0.z = (unsigned int)ga;                         // global_addr[31:0]
    g0.w = (unsigned int)(ga >> 32) | (2u << 30);    // global_addr[56:32] | type=2
    i32x8 g1;
    g1[0] = 0x00010000;                              // data_size=1 (2 bytes)
    g1[1] = (int)((tile_d0 & 0xFFFFu) << 16);        // tensor_dim0[15:0] (== tile_d0)
    g1[2] = (int)(((tile_d0 >> 16) & 0xFFFFu) | ((tile_d1 & 0xFFFFu) << 16)); // td0 hi | td1 lo
    g1[3] = (int)(((tile_d1 >> 16) & 0xFFFFu) | (tile_d0 << 16));             // td1 hi | tile_dim0
    g1[4] = (int)(tile_d1 & 0xFFFFu);                // tile_dim1 | tile_dim2=0
    g1[5] = (int)stride_elems;                       // tensor_dim0_stride[31:0]
    g1[6] = 0;                                       // stride[47:32] | dim1_stride lo
    g1[7] = 0;
    i32x4 z4 = {0, 0, 0, 0};
#if __clang_major__ >= 23
    i32x8 z8 = {0, 0, 0, 0, 0, 0, 0, 0};
    __builtin_amdgcn_tensor_load_to_lds(g0, g1, z4, z4, z8, 0);
#else
    __builtin_amdgcn_tensor_load_to_lds(g0, g1, z4, z4, 0);
#endif
#endif
}

__device__ __forceinline__ unsigned int lds_offset_of(const void* p) {
    return (unsigned int)(unsigned long long)p;      // low 32 bits of shared-aperture addr
}

// Per-wave compute: 32x64 tile per wave of the 128x128 block.
// 8 waves: wave_m = wave&3 (32 rows each), wave_n = wave>>2 (64 cols each).
// Each wave: 2 A-frags x 4 B-frags x 2 K-chunks = 16 WMMAs per 64-wide K-step.
#define GEMM_COMPUTE()                                                          \
    do {                                                                        \
        _Pragma("unroll")                                                       \
        for (int kk = 0; kk < 64; kk += 32) {                                   \
            v16bf a0 = load_fragA(ldsA, wm * 32 + r,      kk, h);               \
            v16bf a1 = load_fragA(ldsA, wm * 32 + 16 + r, kk, h);               \
            _Pragma("unroll")                                                   \
            for (int n = 0; n < 4; ++n) {                                       \
                v16bf bb = load_fragB(ldsB, wn * 64 + n * 16 + r, kk, h);       \
                acc[0][n] = wmma_bf16(a0, bb, acc[0][n]);                       \
                acc[1][n] = wmma_bf16(a1, bb, acc[1][n]);                       \
            }                                                                   \
        }                                                                       \
    } while (0)

// =====================================================================
// Kernel 1: projection  Out[t][e] = sum_d X[t][d] * W[e][d]   (bf16 out)
// Manual staging (needs f32->bf16 conversion in flight).
// =====================================================================
__global__ __launch_bounds__(256) void proj_kernel(const float* __restrict__ X,
                                                   const float* __restrict__ W,
                                                   unsigned short* __restrict__ Out) {
    __shared__ alignas(16) unsigned short ldsA[128 * 64];
    __shared__ alignas(16) unsigned short ldsB[128 * 64];

    const int tid  = threadIdx.x;
    const int wave = tid >> 5, lane = tid & 31;
    const int h = lane >> 4, r = lane & 15;
    const int wm = wave & 3, wn = wave >> 2;
    const int im = blockIdx.x * 128;   // rows over Bs*Ts
    const int jn = blockIdx.y * 128;   // cols over Ds

    v8f acc[2][4] = {};

    for (int k0 = 0; k0 < Ds; k0 += 64) {
#pragma unroll
        for (int it = 0; it < 8; ++it) {         // A: X f32 -> bf16
            int id = tid + it * 256;
            int row = id >> 4, c4 = (id & 15) * 4;
            f32x4 v = *(const f32x4*)(X + (size_t)(im + row) * Ds + k0 + c4);
            unsigned short* dst = &ldsA[row * 64 + c4];
            dst[0] = f2bf(v.x); dst[1] = f2bf(v.y); dst[2] = f2bf(v.z); dst[3] = f2bf(v.w);
        }
#pragma unroll
        for (int it = 0; it < 8; ++it) {         // B: W f32 -> bf16
            int id = tid + it * 256;
            int row = id >> 4, c4 = (id & 15) * 4;
            f32x4 v = *(const f32x4*)(W + (size_t)(jn + row) * Ds + k0 + c4);
            unsigned short* dst = &ldsB[row * 64 + c4];
            dst[0] = f2bf(v.x); dst[1] = f2bf(v.y); dst[2] = f2bf(v.z); dst[3] = f2bf(v.w);
        }
        __syncthreads();
        if (k0 + 64 < Ds) {
            __builtin_prefetch(X + (size_t)(im + (tid & 127)) * Ds + k0 + 64);
            __builtin_prefetch(W + (size_t)(jn + (tid & 127)) * Ds + k0 + 64);
        }
        GEMM_COMPUTE();
        __syncthreads();
    }

#pragma unroll
    for (int mi = 0; mi < 2; ++mi) {
        int row0 = im + wm * 32 + mi * 16;
#pragma unroll
        for (int n = 0; n < 4; ++n)
#pragma unroll
            for (int i = 0; i < 8; ++i)
                Out[(size_t)(row0 + i + 8 * h) * Ds + jn + wn * 64 + n * 16 + r] =
                    f2bf(acc[mi][n][i]);
    }
}

// =====================================================================
// Kernel 2: scores  S[b][i][j] = Q[b][i][:] . K[b][j][:]   (f32 out)
// Q/K tiles staged by the Tensor Data Mover (wave 0 issues, TENSORcnt).
// =====================================================================
__global__ __launch_bounds__(256) void scores_kernel(const unsigned short* __restrict__ Q,
                                                     const unsigned short* __restrict__ Kt,
                                                     float* __restrict__ S) {
    const int b  = blockIdx.z;
    const int im = blockIdx.x * 128;   // query rows i
    const int jn = blockIdx.y * 128;   // key cols j
    if (jn + 128 < im) return;         // whole tile has j < i-1 -> never read

    __shared__ alignas(16) unsigned short ldsA[128 * 64];
    __shared__ alignas(16) unsigned short ldsB[128 * 64];

    const int tid  = threadIdx.x;
    const int wave = tid >> 5, lane = tid & 31;
    const int h = lane >> 4, r = lane & 15;
    const int wm = wave & 3, wn = wave >> 2;

    const unsigned short* Qb = Q  + (size_t)b * Ts * Ds;
    const unsigned short* Kb = Kt + (size_t)b * Ts * Ds;
    float* Sb = S + (size_t)b * Ts * Ts;
    const unsigned int offA = lds_offset_of(ldsA);
    const unsigned int offB = lds_offset_of(ldsB);

    v8f acc[2][4] = {};

    for (int k0 = 0; k0 < Ds; k0 += 64) {
#if HAVE_TDM
        if (wave == 0) {
            tdm_load_2d(offA, Qb + (size_t)im * Ds + k0, 64, 128, Ds);
            tdm_load_2d(offB, Kb + (size_t)jn * Ds + k0, 64, 128, Ds);
            __builtin_amdgcn_s_wait_tensorcnt(0);
        }
#else
#pragma unroll
        for (int it = 0; it < 4; ++it) {
            int id = tid + it * 256;
            int row = id >> 3, q8 = (id & 7) * 8;
            *(u32x4*)(&ldsA[row * 64 + q8]) =
                *(const u32x4*)(Qb + (size_t)(im + row) * Ds + k0 + q8);
            *(u32x4*)(&ldsB[row * 64 + q8]) =
                *(const u32x4*)(Kb + (size_t)(jn + row) * Ds + k0 + q8);
        }
#endif
        __syncthreads();
        GEMM_COMPUTE();
        __syncthreads();
    }

#pragma unroll
    for (int mi = 0; mi < 2; ++mi) {
        int row0 = im + wm * 32 + mi * 16;
#pragma unroll
        for (int n = 0; n < 4; ++n)
#pragma unroll
            for (int i = 0; i < 8; ++i)
                Sb[(size_t)(row0 + i + 8 * h) * Ts + jn + wn * 64 + n * 16 + r] =
                    acc[mi][n][i];
    }
}

// =====================================================================
// Kernel 3: row softmax with mask j >= i-1, scale 1/sqrt(Ds)=1/32.
// Writes bf16 probabilities in-place over the front half of the f32 row
// (zeros for masked j).  One block per (b,i) row.
// =====================================================================
__global__ __launch_bounds__(256) void softmax_kernel(float* __restrict__ S) {
    const int rowid = blockIdx.x;          // 0 .. Bs*Ts-1
    const int i = rowid & (Ts - 1);
    float* Srow = S + (size_t)rowid * Ts;

    __shared__ alignas(16) float buf[Ts];
    __shared__ float red[8];

    const int tid = threadIdx.x;
#pragma unroll
    for (int it = 0; it < 2; ++it) {       // stage row to LDS
        int f = tid + it * 256;
        *(f32x4*)(buf + f * 4) = *(const f32x4*)(Srow + f * 4);
    }
    __syncthreads();

    const int j0 = (i > 0) ? (i - 1) : 0;
    const float scale = 0.03125f;          // 1/sqrt(1024)

    float m = -3.402823e38f;
    for (int j = tid; j < Ts; j += 256)
        if (j >= j0) m = fmaxf(m, buf[j]);
#pragma unroll
    for (int off = 16; off; off >>= 1) m = fmaxf(m, __shfl_xor(m, off));
    if ((tid & 31) == 0) red[tid >> 5] = m;
    __syncthreads();
    float mm = red[0];
#pragma unroll
    for (int k = 1; k < 8; ++k) mm = fmaxf(mm, red[k]);
    __syncthreads();

    float s = 0.f;
    for (int j = tid; j < Ts; j += 256)
        if (j >= j0) s += __expf((buf[j] - mm) * scale);
#pragma unroll
    for (int off = 16; off; off >>= 1) s += __shfl_xor(s, off);
    if ((tid & 31) == 0) red[tid >> 5] = s;
    __syncthreads();
    float tot = 0.f;
#pragma unroll
    for (int k = 0; k < 8; ++k) tot += red[k];
    const float inv = 1.0f / tot;

    unsigned short* Prow = (unsigned short*)Srow;   // bf16 row, stride 2*Ts ushorts
    for (int j = tid; j < Ts; j += 256) {
        float p = (j >= j0) ? __expf((buf[j] - mm) * scale) * inv : 0.f;
        Prow[j] = f2bf(p);
    }
}

// =====================================================================
// Kernel 4: O[b][i][e] = sum_j P[b][i][j] * V[b][j][e]    (f32 out)
// P tile staged by TDM (overlaps the manual V transpose staging).
// Skips K-tiles where P is all-zero (k0+63 < i_min-1).
// =====================================================================
__global__ __launch_bounds__(256) void pv_kernel(const float* __restrict__ Sbase,
                                                 const unsigned short* __restrict__ V,
                                                 float* __restrict__ O) {
    const int b  = blockIdx.z;
    const int im = blockIdx.x * 128;   // rows i
    const int en = blockIdx.y * 128;   // cols e

    __shared__ alignas(16) unsigned short ldsA[128 * 64];
    __shared__ alignas(16) unsigned short ldsB[128 * 64];

    const int tid  = threadIdx.x;
    const int wave = tid >> 5, lane = tid & 31;
    const int h = lane >> 4, r = lane & 15;
    const int wm = wave & 3, wn = wave >> 2;

    const unsigned short* Pb = (const unsigned short*)(Sbase + (size_t)b * Ts * Ts);
    const unsigned short* Vb = V + (size_t)b * Ts * Ds;
    float* Ob = O + (size_t)b * Ts * Ds;
    const unsigned int offA = lds_offset_of(ldsA);

    v8f acc[2][4] = {};
    const int kstart = (im >= 64) ? (im - 64) : 0;   // first tile with any p != 0

    for (int k0 = kstart; k0 < Ts; k0 += 64) {
#if HAVE_TDM
        if (wave == 0)                   // A: P tile [128][64], row stride 2*Ts
            tdm_load_2d(offA, Pb + (size_t)im * (2 * Ts) + k0, 64, 128, 2 * Ts);
#else
#pragma unroll
        for (int it = 0; it < 4; ++it) {
            int id = tid + it * 256;
            int row = id >> 3, q8 = (id & 7) * 8;
            *(u32x4*)(&ldsA[row * 64 + q8]) =
                *(const u32x4*)(Pb + (size_t)(im + row) * (2 * Ts) + k0 + q8);
        }
#endif
#pragma unroll
        for (int it = 0; it < 4; ++it) {     // B: V[k0..k0+63][en..en+127] -> [N][K]
            int id = tid + it * 256;
            int k  = id & 63;                // K within tile
            int c8 = id >> 6;                // 0..15, chunk of 8 e-cols
            union { u32x4 q; unsigned short s[8]; } u;
            u.q = *(const u32x4*)(Vb + (size_t)(k0 + k) * Ds + en + c8 * 8);
#pragma unroll
            for (int x = 0; x < 8; ++x)
                ldsB[(c8 * 8 + x) * 64 + k] = u.s[x];
        }
#if HAVE_TDM
        if (wave == 0) __builtin_amdgcn_s_wait_tensorcnt(0);
#endif
        __syncthreads();
        if (k0 + 64 < Ts)
            __builtin_prefetch(Vb + (size_t)(k0 + 64 + (tid & 63)) * Ds + en + (tid >> 6) * 32);
        GEMM_COMPUTE();
        __syncthreads();
    }

#pragma unroll
    for (int mi = 0; mi < 2; ++mi) {
        int row0 = im + wm * 32 + mi * 16;
#pragma unroll
        for (int n = 0; n < 4; ++n)
#pragma unroll
            for (int i = 0; i < 8; ++i)
                Ob[(size_t)(row0 + i + 8 * h) * Ds + en + wn * 64 + n * 16 + r] =
                    acc[mi][n][i];
    }
}

// =====================================================================
extern "C" void kernel_launch(void* const* d_in, const int* in_sizes, int n_in,
                              void* d_out, int out_size, void* d_ws, size_t ws_size,
                              hipStream_t stream) {
    (void)in_sizes; (void)n_in; (void)out_size; (void)ws_size;
    const float* X  = (const float*)d_in[0];
    const float* Wq = (const float*)d_in[1];
    const float* Wk = (const float*)d_in[2];
    const float* Wv = (const float*)d_in[3];
    float* Out = (float*)d_out;

    const size_t QN = (size_t)Bs * Ts * Ds;          // 8M elements
    unsigned short* Qp = (unsigned short*)d_ws;      // 16 MB
    unsigned short* Kp = Qp + QN;                    // 16 MB
    unsigned short* Vp = Kp + QN;                    // 16 MB
    float* Sp = (float*)(Vp + QN);                   // 64 MB scores / bf16 P

    dim3 gp(64, 8);                                  // (Bs*Ts)/128 x Ds/128
    proj_kernel<<<gp, 256, 0, stream>>>(X, Wq, Qp);
    proj_kernel<<<gp, 256, 0, stream>>>(X, Wk, Kp);
    proj_kernel<<<gp, 256, 0, stream>>>(X, Wv, Vp);

    scores_kernel<<<dim3(Ts / 128, Ts / 128, Bs), 256, 0, stream>>>(Qp, Kp, Sp);
    softmax_kernel<<<dim3(Bs * Ts), 256, 0, stream>>>(Sp);
    pv_kernel<<<dim3(Ts / 128, Ds / 128, Bs), 256, 0, stream>>>(Sp, Vp, Out);
}
